// classifier_27212912787835
// MI455X (gfx1250) — matrix operand order repaired and checked by reading, not compiled, not run on hardware
//
#include <hip/hip_runtime.h>
#include <hip/hip_bf16.h>
#include <math.h>

typedef _Float16 v16h  __attribute__((ext_vector_type(16)));
typedef _Float16 v8h   __attribute__((ext_vector_type(8)));
typedef float    v8f   __attribute__((ext_vector_type(8)));
typedef float    f32x4 __attribute__((ext_vector_type(4)));

#define DDIM 256
#define CDIM 64
#define LDS_PITCH 260   // floats per row; 260%64=4 -> staggers LDS banks across rows

// ---------------- Pass 1: normalize prototypes -> f16 [C][D] row-major ----------------
__global__ __launch_bounds__(64) void proto_norm_kernel(const float* __restrict__ proto,
                                                        _Float16* __restrict__ pB) {
  const int row  = blockIdx.x * 2 + (threadIdx.x >> 5);   // one wave per prototype row
  const int lane = threadIdx.x & 31;
  const float* src = proto + (size_t)row * DDIM + lane * 8;
  f32x4 a = *(const f32x4*)src;
  f32x4 b = *(const f32x4*)(src + 4);
  float ss = a[0]*a[0] + a[1]*a[1] + a[2]*a[2] + a[3]*a[3]
           + b[0]*b[0] + b[1]*b[1] + b[2]*b[2] + b[3]*b[3];
  #pragma unroll
  for (int off = 16; off > 0; off >>= 1) ss += __shfl_xor(ss, off, 32);
  const float s = 1.0f / fmaxf(sqrtf(ss), 1e-12f);
  v8h o;
  #pragma unroll
  for (int i = 0; i < 4; ++i) { o[i] = (_Float16)(a[i] * s); o[4 + i] = (_Float16)(b[i] * s); }
  *(v8h*)(pB + (size_t)row * DDIM + lane * 8) = o;
}

// ---------------- Pass 2: normalize features + WMMA GEMM + distance epilogue ----------------
__global__ __launch_bounds__(64) void logits_kernel(const float* __restrict__ feat,
                                                    const _Float16* __restrict__ pB,
                                                    const float* __restrict__ dscale,
                                                    float* __restrict__ out) {
  __shared__ float lds[2 * 16 * LDS_PITCH];
  const int lane = threadIdx.x & 31;
  const int wave = threadIdx.x >> 5;
  const int row0 = blockIdx.x * 32 + wave * 16;      // this wave's 16-row tile
  float* tile = lds + wave * 16 * LDS_PITCH;

  // Phase A: async-copy 16x256 f32 tile global -> LDS (no VGPR staging; ASYNCcnt path).
  // NOTE: inst_offset applies to BOTH global and LDS addresses, so per-row bases are
  // computed in registers (LDS pitch 260 != global stride 256); the +512B half-row step
  // is identical on both sides and uses the offset field.
  {
    const unsigned long long gbase =
        (unsigned long long)(const void*)(feat + (size_t)row0 * DDIM + lane * 4);
    const unsigned lbase = (unsigned)(size_t)(const void*)&tile[lane * 4]; // AS(3) byte offset
    #pragma unroll
    for (int rr = 0; rr < 16; ++rr) {
      const unsigned long long g0 = gbase + (unsigned)(rr * DDIM * 4);
      const unsigned           l0 = lbase + (unsigned)(rr * LDS_PITCH * 4);
      asm volatile("global_load_async_to_lds_b128 %0, %1, off"
                   :: "v"(l0), "v"(g0) : "memory");
      asm volatile("global_load_async_to_lds_b128 %0, %1, off offset:512"
                   :: "v"(l0), "v"(g0) : "memory");
    }
    asm volatile("s_wait_asynccnt 0x0" ::: "memory");
  }

  // Phase B: per-row sum of squares, 2 lanes per row
  const int r = lane & 15;          // row within tile (also N index for B/C frags)
  const int h = lane >> 4;          // lane-half select
  const int rbase = r * LDS_PITCH;
  float ss = 0.f;
  #pragma unroll
  for (int j = 0; j < 32; ++j) {
    f32x4 v = *(const f32x4*)&tile[rbase + h * 128 + j * 4];
    ss += v[0]*v[0] + v[1]*v[1] + v[2]*v[2] + v[3]*v[3];
  }
  ss += __shfl_xor(ss, 16, 32);
  const float scale = 1.0f / fmaxf(sqrtf(ss), 1e-12f);   // torch F.normalize semantics

  // Phase C: build all 8 A-fragments (16x32 f16, CDNA5 layout):
  // lane half h holds K = k0 + {8h..8h+7} and k0 + {16+8h..16+8h+7}
  v16h afr[8];
  #pragma unroll
  for (int k = 0; k < 8; ++k) {
    const int b2 = rbase + k * 32 + h * 8;
    f32x4 x0 = *(const f32x4*)&tile[b2];
    f32x4 x1 = *(const f32x4*)&tile[b2 + 4];
    f32x4 x2 = *(const f32x4*)&tile[b2 + 16];
    f32x4 x3 = *(const f32x4*)&tile[b2 + 20];
    v16h t;
    #pragma unroll
    for (int i = 0; i < 4; ++i) {
      t[i]      = (_Float16)(x0[i] * scale);
      t[4 + i]  = (_Float16)(x1[i] * scale);
      t[8 + i]  = (_Float16)(x2[i] * scale);
      t[12 + i] = (_Float16)(x3[i] * scale);
    }
    afr[k] = t;
  }

  const float alpha = fabsf(dscale[0]);

  // Phase D: 4 N-tiles x 8 K-steps of v_wmma_f32_16x16x32_f16
  #pragma unroll
  for (int nt = 0; nt < 4; ++nt) {
    v8f acc = {};
    #pragma unroll
    for (int k = 0; k < 8; ++k) {
      // B fragment (32x16 f16): lane (h,r) holds K = k*32 + {8h..8h+7} and k*32 + {16+8h..16+8h+7}, N = r
      const _Float16* bp = pB + (size_t)(nt * 16 + r) * DDIM + k * 32 + h * 8;
      const v8h blo = *(const v8h*)bp;
      const v8h bhi = *(const v8h*)(bp + 16);
      const v16h bf = __builtin_shufflevector(blo, bhi, 0, 1, 2, 3, 4, 5, 6, 7, 8, 9, 10, 11, 12, 13, 14, 15);
      acc = __builtin_amdgcn_wmma_f32_16x16x32_f16(false, afr[k], false, bf,
                                                   (short)0, acc, false, false);
    }
    // C fragment: element j -> (M = 8h + j, N = r)
    #pragma unroll
    for (int j = 0; j < 8; ++j) {
      float sim = acc[j];
      float dd  = sqrtf(fmaxf(2.0f - 2.0f * sim, 0.0f));
      out[(size_t)(row0 + h * 8 + j) * CDIM + nt * 16 + r] = -alpha * dd;
    }
  }
}

extern "C" void kernel_launch(void* const* d_in, const int* in_sizes, int n_in,
                              void* d_out, int out_size, void* d_ws, size_t ws_size,
                              hipStream_t stream) {
  const float* feat   = (const float*)d_in[0];   // [N, 256] f32
  const float* proto  = (const float*)d_in[1];   // [64, 256] f32
  const float* dscale = (const float*)d_in[2];   // [1] f32
  float* out = (float*)d_out;                    // [N, 64] f32
  _Float16* pB = (_Float16*)d_ws;                // 64*256 f16 = 32KB scratch

  const int N = in_sizes[0] / DDIM;              // 262144

  proto_norm_kernel<<<dim3(CDIM / 2), dim3(64), 0, stream>>>(proto, pB);
  logits_kernel<<<dim3(N / 32), dim3(64), 0, stream>>>(feat, pB, dscale, out);
}
